// MultiHeadAttentionWithRope_80848464380534
// MI455X (gfx1250) — matrix operand order repaired and checked by reading, not compiled, not run on hardware
//
#include <hip/hip_runtime.h>
#include <hip/hip_bf16.h>

// ---------------------------------------------------------------------------
// MultiHeadAttentionWithRope for MI455X (gfx1250, wave32, WMMA bf16 path)
//   B=4, S=2048, D=1024, H=16, Dh=64, THETA=1e4, causal.
//   Round 2: attention tiles staged via Tensor Data Mover (tensor_load_to_lds
//   + s_wait_tensorcnt); V kept transposed [b][h][d][s] in global so the TDM
//   delivers the [d][kv] LDS layout WMMA's B-fragments want, no LDS transpose.
// ---------------------------------------------------------------------------

#define D_MODEL 1024
#define SEQ     2048
#define NHEAD   16
#define HDIM    64

typedef __attribute__((ext_vector_type(16))) __bf16 v16bf;
typedef __attribute__((ext_vector_type(8)))  float  v8f;
typedef __attribute__((ext_vector_type(4)))  unsigned int u32x4;
typedef __attribute__((ext_vector_type(8)))  int i32x8;
typedef __attribute__((ext_vector_type(4)))  int i32x4;

#if defined(__HIP_DEVICE_COMPILE__) && __has_builtin(__builtin_amdgcn_tensor_load_to_lds)
#define USE_TDM 1
#endif

__device__ inline v8f vzero8() {
  v8f z = {0.f, 0.f, 0.f, 0.f, 0.f, 0.f, 0.f, 0.f};
  return z;
}

__device__ inline v8f wmma_bf16(v16bf a, v16bf b, v8f c) {
  // D = A(16x32 bf16) * B(32x16 bf16) + C(16x16 f32)
  return __builtin_amdgcn_wmma_f32_16x16x32_bf16(
      /*neg_a=*/false, a, /*neg_b=*/false, b,
      /*c_mod=*/(short)0, c, /*reuse_a=*/false, /*reuse_b=*/false);
}

// A-matrix fragment (16x32, 16-bit) from row-major LDS tile [m][k], stride ld.
__device__ inline v16bf load_frag_a(const __bf16* base, int ld, int k0, int lane) {
  int m = lane & 15, hi = lane >> 4;
  const __bf16* row = base + m * ld + k0 + 8 * hi;
  v16bf a;
#pragma unroll
  for (int v = 0; v < 8; ++v) {
    int k = (v < 4) ? (2 * v) : (16 + 2 * (v - 4));
    a[2 * v]     = row[k];
    a[2 * v + 1] = row[k + 1];
  }
  return a;
}

// B-matrix fragment (32x16, 16-bit) from LDS stored B^T-style:
// element B[k][n] lives at baseT[n*ld + k] (k contiguous -> b128 loads).
__device__ inline v16bf load_frag_b(const __bf16* baseT, int ld, int n0, int k0, int lane) {
  int n = lane & 15, hi = lane >> 4;
  v16bf b;
#pragma unroll
  for (int v = 0; v < 8; ++v) {
    int k = k0 + 8 * hi + ((v < 4) ? (2 * v) : (16 + 2 * (v - 4)));
    const __bf16* p = baseT + (n0 + n) * ld + k;
    b[2 * v]     = p[0];
    b[2 * v + 1] = p[1];
  }
  return b;
}

#ifdef USE_TDM
// Issue one TDM 2-D tile load: global (bf16, row stride strideElems) -> LDS.
// D# layout per CDNA5 ISA (data_size=2B, type=2).
__device__ __attribute__((always_inline)) inline
void tdm_load_2d(const void* gaddr, unsigned ldsOff,
                 unsigned tileX, unsigned tileY,
                 unsigned tensorX, unsigned tensorY, unsigned strideElems) {
  unsigned long long ga = (unsigned long long)gaddr;
  u32x4 g0;
  g0[0] = 1u;                                            // count=1, user mode
  g0[1] = ldsOff;                                        // lds_addr
  g0[2] = (unsigned)(ga & 0xffffffffu);                  // global_addr[31:0]
  g0[3] = (unsigned)((ga >> 32) & 0x01ffffffu) | 0x80000000u; // [56:32] | type=2
  i32x8 g1;
  g1[0] = (int)(1u << 16);                               // data_size=1 (2 bytes)
  g1[1] = (int)((tensorX & 0xffffu) << 16);              // tensor_dim0 lo16
  g1[2] = (int)((tensorX >> 16) | ((tensorY & 0xffffu) << 16));
  g1[3] = (int)((tensorY >> 16) | (tileX << 16));        // tile_dim0
  g1[4] = (int)tileY;                                    // tile_dim1 (tile_dim2=0)
  g1[5] = (int)strideElems;                              // tensor_dim0_stride lo32
  g1[6] = 0;
  g1[7] = 0;
  i32x4 z4 = {0, 0, 0, 0};
#if __clang_major__ >= 23
  i32x8 z8 = {0, 0, 0, 0, 0, 0, 0, 0};
  __builtin_amdgcn_tensor_load_to_lds(g0, g1, z4, z4, z8, 0);
#else
  __builtin_amdgcn_tensor_load_to_lds(g0, g1, z4, z4, 0);
#endif
}
__device__ inline unsigned lds_off(const void* p) {
  return (unsigned)(unsigned long long)p;
}
#endif

// ---------------------------------------------------------------------------
// Kernel 1: projection GEMM  out = rope?( X @ W + bias ), bf16 head layout.
//   applyRope=1 -> out[b][h][s][dh] with RoPE;  applyRope=0 -> V transposed
//   out[b][h][dh][s] (so attention's TDM delivers [d][kv] tiles directly).
//   grid = (16 heads, 128 m-tiles), block = 128 (4 waves).
// ---------------------------------------------------------------------------
__global__ __launch_bounds__(128)
void mha_qkv_proj_rope(const float* __restrict__ X, const float* __restrict__ W,
                       const float* __restrict__ bias, const int* __restrict__ pos,
                       __bf16* __restrict__ out, int applyRope) {
  __shared__ __bf16 As[64 * 32];   // [m][k]
  __shared__ __bf16 Bs[64 * 32];   // transposed: [n][k]
  __shared__ float  Cs[64 * 64];   // fp32 epilogue tile

  const int tid = threadIdx.x, wv = tid >> 5, lane = tid & 31;
  const int n = lane & 15, hi = lane >> 4;
  const int h  = blockIdx.x;
  const int n0 = h * 64;
  const int m0 = blockIdx.y * 64;

  v8f acc[4];
#pragma unroll
  for (int t = 0; t < 4; ++t) acc[t] = vzero8();

  for (int kc = 0; kc < D_MODEL; kc += 32) {
    __syncthreads();
    for (int i = tid; i < 64 * 32; i += 128) {           // A tile 64x32 fp32->bf16
      int r = i >> 5, c = i & 31;
      As[i] = (__bf16)X[(m0 + r) * D_MODEL + kc + c];
      if (kc + 32 < D_MODEL)
        __builtin_prefetch(&X[(m0 + r) * D_MODEL + kc + 32 + c], 0, 0);
    }
    for (int i = tid; i < 32 * 64; i += 128) {           // B tile 32x64 -> [n][k]
      int r = i >> 6, c = i & 63;
      Bs[c * 32 + r] = (__bf16)W[(kc + r) * D_MODEL + n0 + c];
    }
    __syncthreads();
    v16bf af = load_frag_a(As + 16 * wv * 32, 32, 0, lane);
#pragma unroll
    for (int t = 0; t < 4; ++t) {
      v16bf bf = load_frag_b(Bs, 32, t * 16, 0, lane);
      acc[t] = wmma_bf16(af, bf, acc[t]);
    }
  }

  // C fragment -> LDS (with bias), then epilogue.
#pragma unroll
  for (int t = 0; t < 4; ++t)
#pragma unroll
    for (int v = 0; v < 8; ++v) {
      int m = v + 8 * hi;
      Cs[(16 * wv + m) * 64 + t * 16 + n] = acc[t][v] + bias[n0 + t * 16 + n];
    }
  __syncthreads();

  if (applyRope) {
    for (int p = tid; p < 64 * 32; p += 128) {           // (row, pair) items
      int r = p >> 5, c2 = p & 31;
      int rowG = m0 + r, b = rowG >> 11, s = rowG & (SEQ - 1);
      float x1 = Cs[r * 64 + 2 * c2];
      float x2 = Cs[r * 64 + 2 * c2 + 1];
      float P  = (float)pos[b * SEQ + s];
      float ang = P * __expf((float)c2 * (-9.210340371976184f / 32.0f));
      float sn, cs;
      __sincosf(ang, &sn, &cs);
      int o = ((b * NHEAD + h) * SEQ + s) * HDIM;
      out[o + 2 * c2]     = (__bf16)(x1 * cs - x2 * sn);
      out[o + 2 * c2 + 1] = (__bf16)(x1 * sn + x2 * cs);
    }
  } else {
    // V: transposed global layout [b][h][dh][s]; consecutive tid -> consecutive s.
    for (int i = tid; i < 64 * 64; i += 128) {
      int sl = i & 63, c = i >> 6;
      int rowG = m0 + sl, b = rowG >> 11, s = rowG & (SEQ - 1);
      out[((b * NHEAD + h) * HDIM + c) * SEQ + s] = (__bf16)Cs[sl * 64 + c];
    }
  }
}

// ---------------------------------------------------------------------------
// Kernel 2: causal flash attention; tiles staged by the Tensor Data Mover.
//   grid = (S/64 = 32, B*H = 64), block = 128 (4 waves; wave owns 16 q rows).
// ---------------------------------------------------------------------------
__global__ __launch_bounds__(128)
void mha_flash_attn(const __bf16* __restrict__ Qh, const __bf16* __restrict__ Kh,
                    const __bf16* __restrict__ Vt, float* __restrict__ Ob) {
  __shared__ __bf16 Qs[64 * 64];   // [m][d]
  __shared__ __bf16 Ks[64 * 64];   // [kv][d]  (n=kv, k=d for QK^T)
  __shared__ __bf16 Vs[64 * 64];   // [d][kv]  (n=d, k=kv for PV) -- direct from TDM
  __shared__ __bf16 Ps[64 * 64];   // per-wave 16x64 probability tiles

  const int tid = threadIdx.x, wv = tid >> 5, lane = tid & 31;
  const int n = lane & 15, hi = lane >> 4;
  const int qt = blockIdx.x, bh = blockIdx.y;
  const int b = bh >> 4, h = bh & 15;
  const int q0 = qt * 64;
  const float NEG = -__builtin_inff();

  const __bf16* qbase = Qh + (bh * SEQ + q0) * HDIM;
  const __bf16* khead = Kh + bh * SEQ * HDIM;            // [s][d]
  const __bf16* vhead = Vt + bh * HDIM * SEQ;            // [d][s]

#ifdef USE_TDM
  if (wv == 0) {
    tdm_load_2d(qbase, lds_off(Qs), 4096, 1, 4096, 1, 4096);   // 1-D 8KB
    __builtin_amdgcn_s_wait_tensorcnt(0);
  }
#else
  for (int i = tid; i < 64 * 64; i += 128) Qs[i] = qbase[i];
#endif
  __syncthreads();

  v16bf qa0 = load_frag_a(Qs + 16 * wv * 64, 64, 0, lane);
  v16bf qa1 = load_frag_a(Qs + 16 * wv * 64, 64, 32, lane);

  v8f acc[4];
#pragma unroll
  for (int t = 0; t < 4; ++t) acc[t] = vzero8();
  float Mrow[8], Lrow[8];
#pragma unroll
  for (int v = 0; v < 8; ++v) { Mrow[v] = NEG; Lrow[v] = 0.f; }

  for (int kc = 0; kc <= qt; ++kc) {
    __syncthreads();                                     // tiles free to overwrite
#ifdef USE_TDM
    if (wv == 0) {
      // K tile: 64 rows of [s][d], contiguous 8KB.
      tdm_load_2d(khead + kc * 64 * HDIM, lds_off(Ks), 4096, 1, 4096, 1, 4096);
      // V tile: [d=0..63][s=kc*64..+63] from [d][s] with row stride SEQ.
      tdm_load_2d(vhead + kc * 64, lds_off(Vs), 64, 64, SEQ, HDIM, SEQ);
      __builtin_amdgcn_s_wait_tensorcnt(0);
    }
#else
    {
      const __bf16* kb = khead + kc * 64 * HDIM;
      const __bf16* vb = vhead + kc * 64;
      for (int i = tid; i < 64 * 64; i += 128) {
        int r = i >> 6, c = i & 63;
        Ks[i] = kb[i];
        Vs[i] = vb[r * SEQ + c];
      }
    }
#endif
    __syncthreads();

    // S = Q K^T / sqrt(Dh)
    v8f sc[4];
#pragma unroll
    for (int t = 0; t < 4; ++t) {
      sc[t] = vzero8();
      sc[t] = wmma_bf16(qa0, load_frag_b(Ks, 64, t * 16, 0, lane), sc[t]);
      sc[t] = wmma_bf16(qa1, load_frag_b(Ks, 64, t * 16, 32, lane), sc[t]);
    }
    const bool lastc = (kc == qt);
#pragma unroll
    for (int t = 0; t < 4; ++t)
#pragma unroll
      for (int v = 0; v < 8; ++v) {
        float x = sc[t][v] * 0.125f;
        if (lastc && (t * 16 + n > 16 * wv + v + 8 * hi)) x = NEG;
        sc[t][v] = x;
      }

    // Online softmax per row (rows live in 16-lane halves; xor 1/2/4/8 stays inside).
#pragma unroll
    for (int v = 0; v < 8; ++v) {
      float rm = NEG;
#pragma unroll
      for (int t = 0; t < 4; ++t) rm = fmaxf(rm, sc[t][v]);
#pragma unroll
      for (int off = 8; off >= 1; off >>= 1) rm = fmaxf(rm, __shfl_xor(rm, off, 32));
      float mNew = fmaxf(Mrow[v], rm);
      float al   = __expf(Mrow[v] - mNew);
      Mrow[v]    = mNew;
      float rs = 0.f;
#pragma unroll
      for (int t = 0; t < 4; ++t) {
        float p = __expf(sc[t][v] - mNew);
        sc[t][v] = p;
        rs += p;
      }
#pragma unroll
      for (int off = 8; off >= 1; off >>= 1) rs += __shfl_xor(rs, off, 32);
      Lrow[v] = Lrow[v] * al + rs;
#pragma unroll
      for (int t = 0; t < 4; ++t) acc[t][v] *= al;
      int m = v + 8 * hi;
#pragma unroll
      for (int t = 0; t < 4; ++t)
        Ps[(16 * wv + m) * 64 + t * 16 + n] = (__bf16)sc[t][v];
    }
    __syncthreads();

    // O += P @ V
    v16bf pa0 = load_frag_a(Ps + 16 * wv * 64, 64, 0, lane);
    v16bf pa1 = load_frag_a(Ps + 16 * wv * 64, 64, 32, lane);
#pragma unroll
    for (int t = 0; t < 4; ++t) {
      acc[t] = wmma_bf16(pa0, load_frag_b(Vs, 64, t * 16, 0, lane), acc[t]);
      acc[t] = wmma_bf16(pa1, load_frag_b(Vs, 64, t * 16, 32, lane), acc[t]);
    }
  }

  // Epilogue: O / L -> head-merged fp32 [b][s][h*64+d]
#pragma unroll
  for (int t = 0; t < 4; ++t)
#pragma unroll
    for (int v = 0; v < 8; ++v) {
      int m = v + 8 * hi;
      int srow = q0 + 16 * wv + m;
      Ob[(b * SEQ + srow) * D_MODEL + h * HDIM + t * 16 + n] = acc[t][v] / Lrow[v];
    }
}

// ---------------------------------------------------------------------------
// Kernel 3: output projection  Y = O @ w_o + b_o   (fp32 out)
// ---------------------------------------------------------------------------
__global__ __launch_bounds__(128)
void mha_out_proj(const float* __restrict__ A, const float* __restrict__ W,
                  const float* __restrict__ bias, float* __restrict__ Y) {
  __shared__ __bf16 As[64 * 32];
  __shared__ __bf16 Bs[64 * 32];   // transposed [n][k]

  const int tid = threadIdx.x, wv = tid >> 5, lane = tid & 31;
  const int n = lane & 15, hi = lane >> 4;
  const int n0 = blockIdx.x * 64;
  const int m0 = blockIdx.y * 64;

  v8f acc[4];
#pragma unroll
  for (int t = 0; t < 4; ++t) acc[t] = vzero8();

  for (int kc = 0; kc < D_MODEL; kc += 32) {
    __syncthreads();
    for (int i = tid; i < 64 * 32; i += 128) {
      int r = i >> 5, c = i & 31;
      As[i] = (__bf16)A[(m0 + r) * D_MODEL + kc + c];
    }
    for (int i = tid; i < 32 * 64; i += 128) {
      int r = i >> 6, c = i & 63;
      Bs[c * 32 + r] = (__bf16)W[(kc + r) * D_MODEL + n0 + c];
    }
    __syncthreads();
    v16bf af = load_frag_a(As + 16 * wv * 32, 32, 0, lane);
#pragma unroll
    for (int t = 0; t < 4; ++t)
      acc[t] = wmma_bf16(af, load_frag_b(Bs, 32, t * 16, 0, lane), acc[t]);
  }

#pragma unroll
  for (int t = 0; t < 4; ++t)
#pragma unroll
    for (int v = 0; v < 8; ++v) {
      int m = v + 8 * hi;
      int rowG = m0 + 16 * wv + m;
      int colG = n0 + t * 16 + n;
      Y[rowG * D_MODEL + colG] = acc[t][v] + bias[colG];
    }
}

// ---------------------------------------------------------------------------
extern "C" void kernel_launch(void* const* d_in, const int* in_sizes, int n_in,
                              void* d_out, int out_size, void* d_ws, size_t ws_size,
                              hipStream_t stream) {
  const float* query = (const float*)d_in[0];
  const float* key   = (const float*)d_in[1];
  const float* value = (const float*)d_in[2];
  const int*   pos   = (const int*)d_in[3];
  const float* w_q = (const float*)d_in[4];
  const float* b_q = (const float*)d_in[5];
  const float* w_k = (const float*)d_in[6];
  const float* b_k = (const float*)d_in[7];
  const float* w_v = (const float*)d_in[8];
  const float* b_v = (const float*)d_in[9];
  const float* w_o = (const float*)d_in[10];
  const float* b_o = (const float*)d_in[11];
  float* out = (float*)d_out;

  const size_t NE = (size_t)4 * NHEAD * SEQ * HDIM;   // 8,388,608 elems
  __bf16* Qh = (__bf16*)d_ws;
  __bf16* Kh = Qh + NE;
  __bf16* Vt = Kh + NE;                                // transposed [b][h][d][s]
  float*  Ob = (float*)(Vt + NE);                      // total ws use: 80 MB

  dim3 tb(128);
  dim3 gproj(D_MODEL / 64, (4 * SEQ) / 64);            // (16, 128)
  dim3 gattn(SEQ / 64, 4 * NHEAD);                     // (32, 64)

  mha_qkv_proj_rope<<<gproj, tb, 0, stream>>>(query, w_q, b_q, pos, Qh, 1);
  mha_qkv_proj_rope<<<gproj, tb, 0, stream>>>(key,   w_k, b_k, pos, Kh, 1);
  mha_qkv_proj_rope<<<gproj, tb, 0, stream>>>(value, w_v, b_v, pos, Vt, 0);
  mha_flash_attn<<<gattn, tb, 0, stream>>>(Qh, Kh, Vt, Ob);
  mha_out_proj<<<gproj, tb, 0, stream>>>(Ob, w_o, b_o, out);
}
